// Attention_87325275062680
// MI455X (gfx1250) — compile-verified, matrix operand order checked by reading, not run on hardware
//
#include <hip/hip_runtime.h>
#include <hip/hip_bf16.h>

// ---------------------------------------------------------------------------
// Problem constants (reference: B=4, S=2048, D=2048, H=16, HD=128)
// ---------------------------------------------------------------------------
constexpr int B_  = 4;
constexpr int S_  = 2048;
constexpr int D_  = 2048;
constexpr int H_  = 16;
constexpr int HD_ = 128;
constexpr int MM_ = B_ * S_;          // 8192 rows for the projection GEMMs

typedef __attribute__((ext_vector_type(16))) __bf16 v16bf;
typedef __attribute__((ext_vector_type(8)))  float  v8f;

// Trivially-constructible 16B vector (HIP's uint4 has user-declared ctors,
// which would delete a union's default ctor).
struct alignas(16) U4 { unsigned int x, y, z, w; };
union Frag { U4 q[2]; v16bf v; };     // 32B: one 16-bit WMMA A/B operand per lane

struct alignas(8) BF4 { __bf16 a, b, c, d; };

// ---------------------------------------------------------------------------
// CDNA5 async copy: global -> LDS without a VGPR round trip (ASYNCcnt).
//   dsaddr = LDS byte offset (low 32 bits of the shared-aperture pointer)
//   global address = SGPR base + per-lane 32-bit byte offset   (GVS form)
// ---------------------------------------------------------------------------
__device__ __forceinline__ void async_load_b128(void* lds_dst,
                                                const void* gbase,
                                                unsigned byte_off) {
  unsigned lds = (unsigned)(size_t)lds_dst;   // aperture low bits == LDS offset
  asm volatile("global_load_async_to_lds_b128 %0, %1, %2"
               :: "v"(lds), "v"(byte_off), "s"(gbase)
               : "memory");
}
__device__ __forceinline__ void wait_async0() {
  asm volatile("s_wait_asynccnt 0x0" ::: "memory");
}

// ---------------------------------------------------------------------------
// fp32 -> bf16 conversion (vectorized; n is always a multiple of 4)
// ---------------------------------------------------------------------------
__global__ __launch_bounds__(256) void cvt_f32_bf16(const float* __restrict__ src,
                                                    __bf16* __restrict__ dst,
                                                    size_t nquads) {
  size_t i = (size_t)blockIdx.x * blockDim.x + threadIdx.x;
  if (i >= nquads) return;
  float4 f = ((const float4*)src)[i];
  BF4 o;
  o.a = (__bf16)f.x; o.b = (__bf16)f.y; o.c = (__bf16)f.z; o.d = (__bf16)f.w;
  ((BF4*)dst)[i] = o;
}

// ---------------------------------------------------------------------------
// C = A[M,K] * W[N,K]^T   (bf16 inputs, fp32 accumulate via v_wmma)
// PERMUTE_BHSD=true : store bf16 into [B,H,S,HD] (fused head transpose)
// PERMUTE_BHSD=false: store OutT row-major [M,N]
// Block tile 128x128x32, 256 threads = 8 waves (4x2 wave grid), each wave
// owns a 32x64 sub-tile = 2x4 WMMA tiles. Double-buffered LDS staged with
// global_load_async_to_lds_b128 so the loop body is WMMA + ds_load only.
// ---------------------------------------------------------------------------
template <bool PERMUTE_BHSD, typename OutT>
__global__ __launch_bounds__(256)
void gemm_bt_wmma(const __bf16* __restrict__ A,
                  const __bf16* __restrict__ W,
                  OutT* __restrict__ C,
                  int M, int N, int K) {
  constexpr int BM = 128, BN = 128, BK = 32, PAD = 8;
  __shared__ __align__(16) __bf16 As[2][BM][BK + PAD];
  __shared__ __align__(16) __bf16 Ws[2][BN][BK + PAD];

  const int tid  = threadIdx.x;
  const int lane = tid & 31;
  const int w    = tid >> 5;       // wave 0..7
  const int wr   = w & 3;          // wave row (M direction)
  const int wc   = w >> 2;         // wave col (N direction)
  const int l15  = lane & 15;
  const int kh   = lane >> 4;      // K-half select (WMMA 16-bit operand layout)

  const int m0 = blockIdx.x * BM;
  const int n0 = blockIdx.y * BN;

  // Async-stage one 128x32 A tile + 128x32 W tile into buffer `buf`.
  auto stage = [&](int buf, int k0) {
#pragma unroll
    for (int i = 0; i < 2; ++i) {
      int c  = tid + i * 256;
      int r  = c >> 2;
      int c8 = (c & 3) * 8;
      async_load_b128(&As[buf][r][c8], A,
                      (unsigned)(((size_t)(m0 + r) * K + k0 + c8) * 2));
      async_load_b128(&Ws[buf][r][c8], W,
                      (unsigned)(((size_t)(n0 + r) * K + k0 + c8) * 2));
    }
  };

  v8f acc[2][4];
#pragma unroll
  for (int mi = 0; mi < 2; ++mi)
#pragma unroll
    for (int ni = 0; ni < 4; ++ni)
#pragma unroll
      for (int e = 0; e < 8; ++e) acc[mi][ni][e] = 0.f;

  stage(0, 0);                         // prologue
  int cur = 0;
  for (int k0 = 0; k0 < K; k0 += BK) {
    wait_async0();                     // my portion of tile `cur` landed
    __syncthreads();                   // everyone's portion landed; prev reads done
    if (k0 + BK < K) {
      stage(cur ^ 1, k0 + BK);         // overlap next stage with this compute
      __builtin_prefetch(&A[(size_t)(m0 + (tid & 127)) * K + k0 + 2 * BK], 0, 3);
      __builtin_prefetch(&W[(size_t)(n0 + (tid & 127)) * K + k0 + 2 * BK], 0, 3);
    }

    // Fragment loads per documented 16-bit layout:
    //   lane: row = (lane&15), K chunks at kh*8 and kh*8+16 (contiguous 16B).
    Frag af[2], bfw[4];
#pragma unroll
    for (int mi = 0; mi < 2; ++mi) {
      int r = wr * 32 + mi * 16 + l15;
      af[mi].q[0] = *(const U4*)&As[cur][r][kh * 8];
      af[mi].q[1] = *(const U4*)&As[cur][r][kh * 8 + 16];
    }
#pragma unroll
    for (int ni = 0; ni < 4; ++ni) {
      int r = wc * 64 + ni * 16 + l15;
      bfw[ni].q[0] = *(const U4*)&Ws[cur][r][kh * 8];
      bfw[ni].q[1] = *(const U4*)&Ws[cur][r][kh * 8 + 16];
    }
#pragma unroll
    for (int mi = 0; mi < 2; ++mi)
#pragma unroll
      for (int ni = 0; ni < 4; ++ni)
        acc[mi][ni] = __builtin_amdgcn_wmma_f32_16x16x32_bf16(
            false, af[mi].v, false, bfw[ni].v, (short)0, acc[mi][ni],
            false, false);
    cur ^= 1;
  }

  // Epilogue. C layout: lane holds (M = j + 8*kh, N = lane&15) for VGPR j.
#pragma unroll
  for (int mi = 0; mi < 2; ++mi)
#pragma unroll
    for (int ni = 0; ni < 4; ++ni)
#pragma unroll
      for (int j = 0; j < 8; ++j) {
        int gm = m0 + wr * 32 + mi * 16 + j + 8 * kh;
        int gn = n0 + wc * 64 + ni * 16 + l15;
        float val = acc[mi][ni][j];
        if constexpr (PERMUTE_BHSD) {
          int b = gm >> 11, s = gm & (S_ - 1);
          int h = gn >> 7,  hd = gn & (HD_ - 1);
          ((__bf16*)C)[(((size_t)b * H_ + h) * S_ + s) * HD_ + hd] = (__bf16)val;
        } else {
          C[(size_t)gm * N + gn] = (OutT)val;
        }
      }
}

// ---------------------------------------------------------------------------
// In-place RoPE on a [B,H,S,HD] bf16 tensor. One thread per (even,odd) pair.
// ---------------------------------------------------------------------------
__global__ __launch_bounds__(256)
void rope_inplace(__bf16* __restrict__ t,
                  const float* __restrict__ cs,
                  const float* __restrict__ sn) {
  size_t i = (size_t)blockIdx.x * blockDim.x + threadIdx.x;  // pair index
  constexpr size_t total = (size_t)B_ * H_ * S_ * (HD_ / 2);
  if (i >= total) return;
  int    p    = (int)(i & 63);             // frequency index 0..63
  int    s    = (int)((i >> 6) & (S_ - 1));
  size_t base = (i >> 6) * (size_t)HD_ + 2 * p;
  float tr = (float)t[base], ti = (float)t[base + 1];
  float c  = cs[s * 64 + p], sv = sn[s * 64 + p];
  t[base]     = (__bf16)(tr * c - ti * sv);
  t[base + 1] = (__bf16)(tr * sv + ti * c);
}

// ---------------------------------------------------------------------------
// Flash attention, one block per (q-tile of 128, head, batch).
// 8 waves; wave w owns Q rows [w*16, w*16+16). Online softmax in fp32.
// Q and K tiles staged via async-to-LDS; V is transposed through VGPRs.
// Per 32-key step: 8 QK^T WMMAs + 8 PV WMMAs per wave.
// ---------------------------------------------------------------------------
__global__ __launch_bounds__(256)
void flash_attn_wmma(const __bf16* __restrict__ Q,
                     const __bf16* __restrict__ Kk,
                     const __bf16* __restrict__ V,
                     __bf16* __restrict__ O) {  // [B*S, D] row-major
  constexpr int BQ = 128, BKV = 32, PAD = 8;
  __shared__ __align__(16) __bf16 Qs[BQ][HD_ + PAD];     // 34816 B
  __shared__ __align__(16) __bf16 Ks[BKV][HD_ + PAD];    //  8704 B
  __shared__ __align__(16) __bf16 Vt[HD_][BKV + PAD];    // 10240 B (V transposed)
  __shared__ __align__(16) __bf16 Ps[8][16][BKV + PAD];  // 10240 B (per-wave P)

  const int tid = threadIdx.x, lane = tid & 31, w = tid >> 5;
  const int l15 = lane & 15, kh = lane >> 4;
  const int q0 = blockIdx.x * BQ;
  const int h  = blockIdx.y;
  const int b  = blockIdx.z;
  const size_t headBase = ((size_t)b * H_ + h) * S_ * HD_;
  const __bf16* qhead = Q  + headBase;
  const __bf16* khead = Kk + headBase;

  // Stage the Q tile (128x128 bf16 = 2048 16B chunks) via async copy.
#pragma unroll
  for (int i = 0; i < 8; ++i) {
    int c = tid + i * 256;
    int r = c >> 4, c8 = (c & 15) * 8;
    async_load_b128(&Qs[r][c8], qhead,
                    (unsigned)(((size_t)(q0 + r) * HD_ + c8) * 2));
  }
  wait_async0();
  __syncthreads();

  // Q A-fragments for the 4 K-steps over HD=128, kept in registers.
  Frag qa[4];
  {
    int r = w * 16 + l15;
#pragma unroll
    for (int kk = 0; kk < 4; ++kk) {
      qa[kk].q[0] = *(const U4*)&Qs[r][kk * 32 + kh * 8];
      qa[kk].q[1] = *(const U4*)&Qs[r][kk * 32 + kh * 8 + 16];
    }
  }

  v8f   acc[8];
  float m[8], lsum[8];
#pragma unroll
  for (int i = 0; i < 8; ++i) {
#pragma unroll
    for (int e = 0; e < 8; ++e) acc[i][e] = 0.f;
    m[i] = -1e30f; lsum[i] = 0.f;
  }
  const float scale = 0.08838834764831845f;  // 1/sqrt(128)

  for (int kt = 0; kt < S_; kt += BKV) {
    __syncthreads();  // previous iteration done reading Ks/Vt
    // Stage K tile [32,HD] asynchronously; V tile transposed into Vt[HD][32].
#pragma unroll
    for (int i = 0; i < 2; ++i) {
      int c = tid + i * 256;
      int r = c >> 4, c8 = (c & 15) * 8;
      async_load_b128(&Ks[r][c8], khead,
                      (unsigned)(((size_t)(kt + r) * HD_ + c8) * 2));
      U4 raw = *(const U4*)&V[headBase + (size_t)(kt + r) * HD_ + c8];
      const __bf16* hv = (const __bf16*)&raw;
#pragma unroll
      for (int e = 0; e < 8; ++e) Vt[c8 + e][r] = hv[e];
    }
    wait_async0();
    __syncthreads();

    // S = (Q K^T) * scale : 2 key-tiles of 16, 4 WMMA K-steps each.
    v8f sf[2];
#pragma unroll
    for (int nt = 0; nt < 2; ++nt) {
#pragma unroll
      for (int e = 0; e < 8; ++e) sf[nt][e] = 0.f;
#pragma unroll
      for (int kk = 0; kk < 4; ++kk) {
        Frag bk;
        int r = nt * 16 + l15;  // key row; contiguous in HD
        bk.q[0] = *(const U4*)&Ks[r][kk * 32 + kh * 8];
        bk.q[1] = *(const U4*)&Ks[r][kk * 32 + kh * 8 + 16];
        sf[nt] = __builtin_amdgcn_wmma_f32_16x16x32_bf16(
            false, qa[kk].v, false, bk.v, (short)0, sf[nt], false, false);
      }
    }

    // Online softmax per accumulator row j (global row = j + 8*kh).
#pragma unroll
    for (int j = 0; j < 8; ++j) {
      float v0 = sf[0][j] * scale, v1 = sf[1][j] * scale;
      float tmax = fmaxf(v0, v1);
#pragma unroll
      for (int d = 1; d < 16; d <<= 1) tmax = fmaxf(tmax, __shfl_xor(tmax, d, 32));
      float mnew = fmaxf(m[j], tmax);
      float corr = __expf(m[j] - mnew);
      float p0 = __expf(v0 - mnew);
      float p1 = __expf(v1 - mnew);
      float rs = p0 + p1;
#pragma unroll
      for (int d = 1; d < 16; d <<= 1) rs += __shfl_xor(rs, d, 32);
      lsum[j] = lsum[j] * corr + rs;
      m[j] = mnew;
#pragma unroll
      for (int nt2 = 0; nt2 < 8; ++nt2) acc[nt2][j] *= corr;
      int prow = j + 8 * kh;
      Ps[w][prow][l15]      = (__bf16)p0;   // wave-private: DS is in-order,
      Ps[w][prow][l15 + 16] = (__bf16)p1;   // no barrier needed before reload
    }

    // O += P * V. A = P (16x32), B = V^T rows give contiguous key chunks.
    Frag pa;
    pa.q[0] = *(const U4*)&Ps[w][l15][kh * 8];
    pa.q[1] = *(const U4*)&Ps[w][l15][kh * 8 + 16];
#pragma unroll
    for (int nt2 = 0; nt2 < 8; ++nt2) {
      Frag bv;
      int r = nt2 * 16 + l15;  // hd row of Vt; contiguous keys
      bv.q[0] = *(const U4*)&Vt[r][kh * 8];
      bv.q[1] = *(const U4*)&Vt[r][kh * 8 + 16];
      acc[nt2] = __builtin_amdgcn_wmma_f32_16x16x32_bf16(
          false, pa.v, false, bv.v, (short)0, acc[nt2], false, false);
    }
  }

  // Normalize and store as [B*S, D] bf16 (feeds the Wo GEMM).
#pragma unroll
  for (int nt2 = 0; nt2 < 8; ++nt2)
#pragma unroll
    for (int j = 0; j < 8; ++j) {
      int srow = q0 + w * 16 + j + 8 * kh;
      int col  = h * HD_ + nt2 * 16 + l15;
      float ov = acc[nt2][j] / lsum[j];
      O[((size_t)b * S_ + srow) * (size_t)D_ + col] = (__bf16)ov;
    }
}

// ---------------------------------------------------------------------------
// Host orchestration
// ---------------------------------------------------------------------------
extern "C" void kernel_launch(void* const* d_in, const int* in_sizes, int n_in,
                              void* d_out, int out_size, void* d_ws, size_t ws_size,
                              hipStream_t stream) {
  (void)in_sizes; (void)n_in; (void)out_size;

  const float* x  = (const float*)d_in[0];
  const float* fc = (const float*)d_in[1];
  const float* fs = (const float*)d_in[2];
  const float* wq = (const float*)d_in[3];
  const float* wk = (const float*)d_in[4];
  const float* wv = (const float*)d_in[5];
  const float* wo = (const float*)d_in[6];
  float* out = (float*)d_out;

  // Workspace carve-out (~201.5 MB total).
  char* ws = (char*)d_ws;
  size_t off = 0;
  auto carve = [&](size_t bytes) {
    char* p = ws + off;
    off += (bytes + 255) & ~(size_t)255;
    return p;
  };
  const size_t xElems = (size_t)MM_ * D_;      // 16,777,216
  const size_t wElems = (size_t)D_ * D_;       //  4,194,304
  __bf16* xb    = (__bf16*)carve(xElems * 2);
  __bf16* wqb   = (__bf16*)carve(wElems * 2);
  __bf16* wkb   = (__bf16*)carve(wElems * 2);
  __bf16* wvb   = (__bf16*)carve(wElems * 2);
  __bf16* wob   = (__bf16*)carve(wElems * 2);
  __bf16* qbuf  = (__bf16*)carve(xElems * 2);  // [B,H,S,HD]
  __bf16* kbuf  = (__bf16*)carve(xElems * 2);
  __bf16* vbuf  = (__bf16*)carve(xElems * 2);
  __bf16* attnb = (__bf16*)carve(xElems * 2);  // [B*S, D]
  if (off > ws_size) return;  // workspace too small; nothing safe to do

  // 1) bf16 casts
  {
    size_t nq = xElems / 4;
    cvt_f32_bf16<<<dim3((unsigned)((nq + 255) / 256)), 256, 0, stream>>>(x, xb, nq);
    size_t wqn = wElems / 4;
    dim3 g((unsigned)((wqn + 255) / 256));
    cvt_f32_bf16<<<g, 256, 0, stream>>>(wq, wqb, wqn);
    cvt_f32_bf16<<<g, 256, 0, stream>>>(wk, wkb, wqn);
    cvt_f32_bf16<<<g, 256, 0, stream>>>(wv, wvb, wqn);
    cvt_f32_bf16<<<g, 256, 0, stream>>>(wo, wob, wqn);
  }

  // 2) QKV projections, storing straight into [B,H,S,HD]
  {
    dim3 grid(MM_ / 128, D_ / 128);  // 64 x 16
    gemm_bt_wmma<true, __bf16><<<grid, 256, 0, stream>>>(xb, wqb, qbuf, MM_, D_, D_);
    gemm_bt_wmma<true, __bf16><<<grid, 256, 0, stream>>>(xb, wkb, kbuf, MM_, D_, D_);
    gemm_bt_wmma<true, __bf16><<<grid, 256, 0, stream>>>(xb, wvb, vbuf, MM_, D_, D_);
  }

  // 3) RoPE on Q and K (in place)
  {
    size_t pairs = (size_t)B_ * H_ * S_ * (HD_ / 2);  // 8,388,608
    dim3 g((unsigned)((pairs + 255) / 256));
    rope_inplace<<<g, 256, 0, stream>>>(qbuf, fc, fs);
    rope_inplace<<<g, 256, 0, stream>>>(kbuf, fc, fs);
  }

  // 4) Flash attention -> attnb [B*S, D]
  {
    dim3 grid(S_ / 128, H_, B_);  // 16 x 16 x 4
    flash_attn_wmma<<<grid, 256, 0, stream>>>(qbuf, kbuf, vbuf, attnb);
  }

  // 5) Output projection, fp32 result
  {
    dim3 grid(MM_ / 128, D_ / 128);
    gemm_bt_wmma<false, float><<<grid, 256, 0, stream>>>(attnb, wob, out, MM_, D_, D_);
  }
}
